// ImprovedVietnameseVQAModel_30219389894939
// MI455X (gfx1250) — matrix operand order repaired and checked by reading, not compiled, not run on hardware
//
#include <hip/hip_runtime.h>
#include <math.h>

// ---------------------------------------------------------------------------
// Model dims (compile-time)
// ---------------------------------------------------------------------------
#define BB   8
#define SS   2048
#define HH   1024
#define NHH  8
#define DHH  128
#define FFF  4096
#define H3   (3 * HH)
#define MR   (BB * SS)          // 16384 token rows
#define LN_EPS 1e-5f

typedef __bf16 bf16;
typedef __attribute__((ext_vector_type(16))) __bf16 v16bf;
typedef __attribute__((ext_vector_type(8)))  float  v8f;
typedef __attribute__((ext_vector_type(4)))  unsigned int v4u;
typedef __attribute__((ext_vector_type(8)))  int    v8i;
typedef __attribute__((ext_vector_type(4)))  int    v4i;

union Frag16 { v16bf v; uint4 q[2]; };

__device__ __forceinline__ v8f zero8() {
    v8f z = {0.f, 0.f, 0.f, 0.f, 0.f, 0.f, 0.f, 0.f};
    return z;
}

// LDS generic pointer -> 32-bit LDS byte offset (low 32 bits of flat LDS addr)
__device__ __forceinline__ unsigned lds_off32(const void* p) {
    return (unsigned)(unsigned long long)p;
}

// ---------------------------------------------------------------------------
// CDNA5 async global->LDS copy (ASYNCcnt-tracked); per-lane B128.
// ---------------------------------------------------------------------------
__device__ __forceinline__ void async_b128(void* ldsp, const void* g) {
    unsigned l = lds_off32(ldsp);
    unsigned long long a = (unsigned long long)g;
    asm volatile("global_load_async_to_lds_b128 %0, %1, off" :: "v"(l), "v"(a) : "memory");
}
__device__ __forceinline__ void async_wait0() {
    asm volatile("s_wait_asynccnt 0x0" ::: "memory");
}

// ---------------------------------------------------------------------------
// CDNA5 Tensor Data Mover: DMA a [128 rows x 32 cols] bf16 tile from a
// row-major [Mtot x K] tensor into LDS, with HW padding 16B per 64B row
// -> LDS row stride = 40 halves (bank-conflict-free fragment reads).
// Descriptor per CDNA5 ISA ch.8; 6-arg builtin on this toolchain.
// ---------------------------------------------------------------------------
#if __has_builtin(__builtin_amdgcn_tensor_load_to_lds)
#define HAVE_TDM 1
__device__ __forceinline__ void tdm_load_tile_128x32(unsigned lds_byte_off,
                                                     const void* gaddr,
                                                     unsigned K, unsigned Mtot) {
    unsigned long long ga = (unsigned long long)gaddr;
    v4u g0;
    g0.x = 1u;                                            // count=1, no gather
    g0.y = lds_byte_off;                                  // lds_addr
    g0.z = (unsigned)(ga & 0xFFFFFFFFull);                // global_addr[31:0]
    g0.w = (unsigned)((ga >> 32) & 0x01FFFFFFull)         // global_addr[56:32]
         | 0x80000000u;                                   // type=2 ("image")
    v8i g1;
    g1[0] = (1 << 16)        // data_size = 1 -> 2 bytes
          | (1 << 20)        // pad_enable
          | (3 << 22)        // pad_interval code 3 -> every 64 bytes
          | (3 << 25);       // pad_amount  code 3 -> 4 dwords = 16 bytes
    g1[1] = (int)((K & 0xFFFFu) << 16);                   // tensor_dim0[15:0] @ bits 63:48
    g1[2] = (int)((K >> 16) | ((Mtot & 0xFFFFu) << 16));  // tensor_dim0 hi, tensor_dim1 lo
    g1[3] = (int)(((Mtot >> 16) & 0xFFFFu) | (32u << 16));// tensor_dim1 hi, tile_dim0=32
    g1[4] = 128;                                          // tile_dim1=128, tile_dim2=0
    g1[5] = (int)K;                                       // tensor_dim0_stride[31:0]
    g1[6] = 0;
    g1[7] = 0;
    v4i z4 = {0, 0, 0, 0};
    v8i z8 = {0, 0, 0, 0, 0, 0, 0, 0};
    __builtin_amdgcn_tensor_load_to_lds(g0, g1, z4, z4, z8, 0);
}
#else
#define HAVE_TDM 0
#endif

// ---------------------------------------------------------------------------
// fp32 -> bf16 conversion (grid-stride)
// ---------------------------------------------------------------------------
__global__ void k_cvt_bf16(const float* __restrict__ s, bf16* __restrict__ d, long n) {
    long i = (long)blockIdx.x * blockDim.x + threadIdx.x;
    long stride = (long)gridDim.x * blockDim.x;
    for (; i < n; i += stride) d[i] = (bf16)s[i];
}

__global__ void k_zero3(int* p) {
    if (threadIdx.x < 3) p[threadIdx.x] = 0;
}

__global__ void k_build_idx(const int* __restrict__ mm, int* cnt, int* idx, int M) {
    int i = blockIdx.x * blockDim.x + threadIdx.x;
    if (i < M) {
        int m = mm[i];
        int e = (m == 0) ? 0 : ((m == 1) ? 1 : 2);
        int p = atomicAdd(&cnt[e], 1);
        idx[e * M + p] = i;
    }
}

// ---------------------------------------------------------------------------
// GEMM: C[M,N] = A[M,K](bf16) * W[K,N](bf16) + bias
// tile 128x128x32, 8 waves, wave tile 64x32 (4x2 of 16x16 WMMA)
// Double-buffered software pipeline: stage tile i+1 (TDM / async-to-LDS)
// while computing tile i; one barrier per K-step.
// MODE 0: store bf16            (QKV projection)
// MODE 1: store fp32            (attention out-proj)
// MODE 2: gathered A rows (idx), GELU, store bf16 compact  (expert FFN 1)
// MODE 3: compact A rows,  scatter-store fp32 via idx      (expert FFN 2)
// ---------------------------------------------------------------------------
template <int MODE>
__global__ __launch_bounds__(256)
void k_gemm(const bf16* __restrict__ A, const bf16* __restrict__ W,
            const float* __restrict__ bias,
            float* __restrict__ outf, bf16* __restrict__ outb,
            int N, int K,
            const int* __restrict__ idx, const int* __restrict__ cntp) {
    constexpr bool GATHER = (MODE == 2);
    constexpr bool BOUND  = (MODE == 2 || MODE == 3);

    __shared__ __align__(16) bf16 sA[2][128 * 40];   // [row][k], k-stride 40 halves
    __shared__ __align__(16) bf16 sB[2][128 * 40];   // [n][k] transposed, stride 40

    const long row0 = (long)blockIdx.y * 128;
    const int  col0 = blockIdx.x * 128;

    int cntv = 0;
    if (BOUND) {
        cntv = *cntp;
        if (row0 >= cntv) return;          // uniform across block
    }

    const int tid  = threadIdx.x;
    const int lane = tid & 31;
    const int wave = tid >> 5;
    const int wm   = wave >> 2;            // 0..1
    const int wn   = wave & 3;             // 0..3
    const int half = lane >> 4;
    const int lm   = lane & 15;
    const int s8   = half * 8;

    // loop-invariant A staging coordinates (this thread copies one half-row)
    const int rA    = tid >> 1;
    const int koffA = (tid & 1) * 16;
    long rgA = row0 + rA;
    if constexpr (GATHER) rgA = (rgA < cntv) ? (long)idx[rgA] : 0;
    const bf16* arow = A + rgA * (long)K + koffA;   // + k0 per stage

    v8f acc[4][2];
#pragma unroll
    for (int mt = 0; mt < 4; mt++)
#pragma unroll
        for (int nt = 0; nt < 2; nt++) acc[mt][nt] = zero8();

    // ---- stage one K-tile (A via TDM/async, W transposed manually) ----
    auto stage_tiles = [&](int k0, int buf) {
        if constexpr (GATHER) {
            async_b128(&sA[buf][rA * 40 + koffA],     arow + k0);
            async_b128(&sA[buf][rA * 40 + koffA + 8], arow + k0 + 8);
        } else {
#if HAVE_TDM
            if (wave == 0)
                tdm_load_tile_128x32(lds_off32(&sA[buf][0]),
                                     A + row0 * (long)K + k0,
                                     (unsigned)K, (unsigned)MR);
#else
            async_b128(&sA[buf][rA * 40 + koffA],     arow + k0);
            async_b128(&sA[buf][rA * 40 + koffA + 8], arow + k0 + 8);
#endif
        }
#pragma unroll
        for (int i = tid; i < 512; i += 256) {
            int k  = i >> 4;
            int n8 = (i & 15) << 3;
            uint4 w = *(const uint4*)(W + (long)(k0 + k) * N + col0 + n8);
            bf16 tmp[8];
            *(uint4*)tmp = w;
#pragma unroll
            for (int j = 0; j < 8; j++) sB[buf][(n8 + j) * 40 + k] = tmp[j];
        }
    };
    auto wait_stage = [&]() {
        if constexpr (GATHER) {
            async_wait0();
        } else {
#if HAVE_TDM
            if (wave == 0) __builtin_amdgcn_s_wait_tensorcnt(0);
#else
            async_wait0();
#endif
        }
    };

    const int nsteps = K >> 5;
    stage_tiles(0, 0);            // prologue
    int cur = 0;

    for (int i = 0; i < nsteps; i++) {
        wait_stage();             // finish staging of buffer `cur`
        __syncthreads();          // publish to all waves
        if (i + 1 < nsteps)
            stage_tiles((i + 1) << 5, cur ^ 1);   // overlap with compute below

        // ---- fragments + 8 WMMAs from buffer `cur` ----
        Frag16 af[4];
#pragma unroll
        for (int mt = 0; mt < 4; mt++) {
            const bf16* p = &sA[cur][(wm * 64 + mt * 16 + lm) * 40];
            af[mt].q[0] = *(const uint4*)(p + s8);
            af[mt].q[1] = *(const uint4*)(p + 16 + s8);
        }
        Frag16 bfr[2];
#pragma unroll
        for (int nt = 0; nt < 2; nt++) {
            const bf16* p = &sB[cur][(wn * 32 + nt * 16 + lm) * 40];
            bfr[nt].q[0] = *(const uint4*)(p + s8);
            bfr[nt].q[1] = *(const uint4*)(p + 16 + s8);
        }
#pragma unroll
        for (int mt = 0; mt < 4; mt++)
#pragma unroll
            for (int nt = 0; nt < 2; nt++)
                acc[mt][nt] = __builtin_amdgcn_wmma_f32_16x16x32_bf16(
                    false, af[mt].v, false, bfr[nt].v,
                    (short)0, acc[mt][nt], false, false);
        cur ^= 1;
    }

    // ---- epilogue ----
#pragma unroll
    for (int mt = 0; mt < 4; mt++) {
        long growb = row0 + wm * 64 + mt * 16 + half * 8;
#pragma unroll
        for (int nt = 0; nt < 2; nt++) {
            int gcol = col0 + wn * 32 + nt * 16 + lm;
            float bv = bias[gcol];
#pragma unroll
            for (int r = 0; r < 8; r++) {
                long grow = growb + r;
                float val = acc[mt][nt][r] + bv;
                if constexpr (MODE == 0) {
                    outb[grow * (long)N + gcol] = (bf16)val;
                } else if constexpr (MODE == 1) {
                    outf[grow * (long)N + gcol] = val;
                } else if constexpr (MODE == 2) {
                    if (grow < cntv) {
                        float g = 0.5f * val * (1.0f + erff(val * 0.70710678118f));
                        outb[grow * (long)N + gcol] = (bf16)g;
                    }
                } else {  // MODE 3
                    if (grow < cntv)
                        outf[(long)idx[grow] * N + gcol] = val;
                }
            }
        }
    }
}

// ---------------------------------------------------------------------------
// Flash attention, all-WMMA, double-buffered K/V staging.
// Block = (b, h, 128 q rows), 8 waves x 16 q rows.
// ---------------------------------------------------------------------------
__global__ __launch_bounds__(256)
void k_attn(const bf16* __restrict__ qkv, const unsigned char* __restrict__ amask,
            bf16* __restrict__ ctx) {
    __shared__ __align__(16) bf16 sK[2][32 * 136];   // [key][dh], stride 136
    __shared__ __align__(16) bf16 sV[2][128 * 40];   // [dh][key] transposed, stride 40
    __shared__ __align__(16) bf16 sP[8][16 * 40];    // per-wave P tile

    const int b  = blockIdx.z;
    const int h  = blockIdx.y;
    const int q0 = blockIdx.x * 128;
    const int tid = threadIdx.x, lane = tid & 31, wave = tid >> 5;
    const int half = lane >> 4, lm = lane & 15, s8 = half * 8;
    const long rowbase = (long)b * SS;

    // resident Q fragments (16 rows x 128 dh = 4 chunks of 16x32)
    Frag16 qf[4];
    {
        long qrow = rowbase + q0 + wave * 16 + lm;
        const bf16* qp = qkv + qrow * H3 + h * DHH;
#pragma unroll
        for (int c = 0; c < 4; c++) {
            qf[c].q[0] = *(const uint4*)(qp + c * 32 + s8);
            qf[c].q[1] = *(const uint4*)(qp + c * 32 + 16 + s8);
        }
    }

    v8f o[8];
    float m_run[8], l_run[8];
#pragma unroll
    for (int dt = 0; dt < 8; dt++) o[dt] = zero8();
#pragma unroll
    for (int r = 0; r < 8; r++) { m_run[r] = -1e30f; l_run[r] = 0.f; }

    const float scale = 0.08838834764831845f;   // 1/sqrt(128)

    // stage K (async-to-LDS) and V (transposed, manual) for one 32-key chunk
    auto stage_kv = [&](int kc, int buf) {
#pragma unroll
        for (int i = tid; i < 512; i += 256) {
            int key = i >> 4, d8 = (i & 15) << 3;
            async_b128(&sK[buf][key * 136 + d8],
                       qkv + (rowbase + kc + key) * H3 + HH + h * DHH + d8);
        }
#pragma unroll
        for (int i = tid; i < 512; i += 256) {
            int key = i >> 4, d8 = (i & 15) << 3;
            uint4 w = *(const uint4*)(qkv + (rowbase + kc + key) * H3 + 2 * HH + h * DHH + d8);
            bf16 tmp[8];
            *(uint4*)tmp = w;
#pragma unroll
            for (int j = 0; j < 8; j++) sV[buf][(d8 + j) * 40 + key] = tmp[j];
        }
    };

    stage_kv(0, 0);               // prologue
    int cur = 0;

    for (int kc = 0; kc < SS; kc += 32) {
        async_wait0();            // finish staging of buffer `cur`
        __syncthreads();
        if (kc + 32 < SS)
            stage_kv(kc + 32, cur ^ 1);   // overlap with compute below

        // scores: two 16x16 tiles, K-dim = 128 over 4 WMMAs each
        v8f sc[2];
#pragma unroll
        for (int nt = 0; nt < 2; nt++) {
            sc[nt] = zero8();
#pragma unroll
            for (int c = 0; c < 4; c++) {
                Frag16 kf;
                const bf16* kp = &sK[cur][(nt * 16 + lm) * 136 + c * 32];
                kf.q[0] = *(const uint4*)(kp + s8);
                kf.q[1] = *(const uint4*)(kp + 16 + s8);
                sc[nt] = __builtin_amdgcn_wmma_f32_16x16x32_bf16(
                    false, qf[c].v, false, kf.v, (short)0, sc[nt], false, false);
            }
        }

        unsigned char mk0 = amask[rowbase + kc + lm];
        unsigned char mk1 = amask[rowbase + kc + 16 + lm];

        float p0[8], p1[8];
#pragma unroll
        for (int r = 0; r < 8; r++) {
            float s0 = sc[0][r] * scale; if (!mk0) s0 = -1e30f;
            float s1 = sc[1][r] * scale; if (!mk1) s1 = -1e30f;
            float mx = fmaxf(s0, s1);
#pragma unroll
            for (int off = 1; off < 16; off <<= 1)
                mx = fmaxf(mx, __shfl_xor(mx, off, 32));
            float mnew = fmaxf(m_run[r], mx);
            float corr = __expf(m_run[r] - mnew);
            float e0 = __expf(s0 - mnew), e1 = __expf(s1 - mnew);
            float sum = e0 + e1;
#pragma unroll
            for (int off = 1; off < 16; off <<= 1)
                sum += __shfl_xor(sum, off, 32);
            l_run[r] = l_run[r] * corr + sum;
            m_run[r] = mnew;
            p0[r] = e0; p1[r] = e1;
#pragma unroll
            for (int dt = 0; dt < 8; dt++) o[dt][r] *= corr;
        }

        // C-layout -> A-layout via per-wave LDS round trip
        bf16* pw = &sP[wave][0];
#pragma unroll
        for (int r = 0; r < 8; r++) {
            pw[(r + half * 8) * 40 + lm]      = (bf16)p0[r];
            pw[(r + half * 8) * 40 + 16 + lm] = (bf16)p1[r];
        }
        Frag16 pf;
        const bf16* pr = &sP[wave][lm * 40];
        pf.q[0] = *(const uint4*)(pr + s8);
        pf.q[1] = *(const uint4*)(pr + 16 + s8);

        // O += P(16x32) * V(32x128)
#pragma unroll
        for (int dt = 0; dt < 8; dt++) {
            Frag16 vf;
            const bf16* vp = &sV[cur][(dt * 16 + lm) * 40];
            vf.q[0] = *(const uint4*)(vp + s8);
            vf.q[1] = *(const uint4*)(vp + 16 + s8);
            o[dt] = __builtin_amdgcn_wmma_f32_16x16x32_bf16(
                false, pf.v, false, vf.v, (short)0, o[dt], false, false);
        }
        cur ^= 1;
    }

    float inv[8];
#pragma unroll
    for (int r = 0; r < 8; r++) inv[r] = (l_run[r] > 0.f) ? 1.f / l_run[r] : 0.f;

    long qrow0 = rowbase + q0 + wave * 16;
#pragma unroll
    for (int dt = 0; dt < 8; dt++)
#pragma unroll
        for (int r = 0; r < 8; r++)
            ctx[(qrow0 + r + half * 8) * (long)HH + h * DHH + dt * 16 + lm] =
                (bf16)(o[dt][r] * inv[r]);
}

// ---------------------------------------------------------------------------
// out = LN(x + y) * g + b ; one block per row of H=1024, 256 threads x 4
// ---------------------------------------------------------------------------
__global__ __launch_bounds__(256)
void k_add_ln(const float* __restrict__ x, const float* __restrict__ y,
              const float* __restrict__ g, const float* __restrict__ bta,
              float* __restrict__ outf, bf16* __restrict__ outb) {
    __shared__ float rs[256], rs2[256];
    long row = blockIdx.x;
    const float* xr = x + row * HH;
    const float* yr = y + row * HH;
    int t = threadIdx.x;
    float v[4], s = 0.f, s2 = 0.f;
#pragma unroll
    for (int i = 0; i < 4; i++) {
        float a = xr[t + i * 256] + yr[t + i * 256];
        v[i] = a; s += a; s2 += a * a;
    }
    rs[t] = s; rs2[t] = s2;
    __syncthreads();
    for (int off = 128; off > 0; off >>= 1) {
        if (t < off) { rs[t] += rs[t + off]; rs2[t] += rs2[t + off]; }
        __syncthreads();
    }
    float mean = rs[0] * (1.f / HH);
    float var  = rs2[0] * (1.f / HH) - mean * mean;
    float rstd = rsqrtf(var + LN_EPS);
#pragma unroll
    for (int i = 0; i < 4; i++) {
        int c = t + i * 256;
        float o = (v[i] - mean) * rstd * g[c] + bta[c];
        outf[row * HH + c] = o;
        if (outb) outb[row * HH + c] = (bf16)o;
    }
}

// ---------------------------------------------------------------------------
// launch
// ---------------------------------------------------------------------------
extern "C" void kernel_launch(void* const* d_in, const int* in_sizes, int n_in,
                              void* d_out, int out_size, void* d_ws, size_t ws_size,
                              hipStream_t stream) {
    (void)in_sizes; (void)n_in; (void)out_size; (void)ws_size;

    const float* x        = (const float*)d_in[0];
    const int*   mmod     = (const int*)d_in[1];
    const unsigned char* amask = (const unsigned char*)d_in[2];
    const float* Wqkv = (const float*)d_in[3];
    const float* bqkv = (const float*)d_in[4];
    const float* Wout = (const float*)d_in[5];
    const float* bout = (const float*)d_in[6];
    const float* ln1g = (const float*)d_in[7];
    const float* ln1b = (const float*)d_in[8];
    const float* W1s[3] = {(const float*)d_in[9],  (const float*)d_in[13], (const float*)d_in[17]};
    const float* b1s[3] = {(const float*)d_in[10], (const float*)d_in[14], (const float*)d_in[18]};
    const float* W2s[3] = {(const float*)d_in[11], (const float*)d_in[15], (const float*)d_in[19]};
    const float* b2s[3] = {(const float*)d_in[12], (const float*)d_in[16], (const float*)d_in[20]};
    const float* ln2g = (const float*)d_in[21];
    const float* ln2b = (const float*)d_in[22];

    // workspace carve
    size_t off = 0;
    char* base = (char*)d_ws;
    auto carve = [&](size_t bytes) -> void* {
        void* p = base + off;
        off += (bytes + 255) & ~(size_t)255;
        return p;
    };
    const long M = MR;
    bf16* xb    = (bf16*)carve((size_t)M * HH * 2);
    bf16* wqkvb = (bf16*)carve((size_t)HH * H3 * 2);
    bf16* woutb = (bf16*)carve((size_t)HH * HH * 2);
    bf16* w1b[3], *w2b[3];
    for (int e = 0; e < 3; e++) {
        w1b[e] = (bf16*)carve((size_t)HH * FFF * 2);
        w2b[e] = (bf16*)carve((size_t)FFF * HH * 2);
    }
    bf16*  qkvb  = (bf16*)carve((size_t)M * H3 * 2);
    bf16*  ctxb  = (bf16*)carve((size_t)M * HH * 2);
    float* attnf = (float*)carve((size_t)M * HH * 4);
    float* x1f   = (float*)carve((size_t)M * HH * 4);
    bf16*  x1b   = (bf16*)carve((size_t)M * HH * 2);
    bf16*  hb    = (bf16*)carve((size_t)M * FFF * 2);
    float* eoutf = (float*)carve((size_t)M * HH * 4);
    int*   idx   = (int*)carve((size_t)3 * M * 4);
    int*   cnt   = (int*)carve(256);

    // fp32 -> bf16 conversions
    auto cvt = [&](const float* s, bf16* d, long n) {
        k_cvt_bf16<<<1024, 256, 0, stream>>>(s, d, n);
    };
    cvt(x, xb, M * HH);
    cvt(Wqkv, wqkvb, (long)HH * H3);
    cvt(Wout, woutb, (long)HH * HH);
    for (int e = 0; e < 3; e++) {
        cvt(W1s[e], w1b[e], (long)HH * FFF);
        cvt(W2s[e], w2b[e], (long)FFF * HH);
    }

    // QKV projection: [M,H] x [H,3H] -> bf16 [M,3H]
    k_gemm<0><<<dim3(H3 / 128, M / 128), 256, 0, stream>>>(
        xb, wqkvb, bqkv, nullptr, qkvb, H3, HH, nullptr, nullptr);

    // attention -> ctx bf16 [M,H]
    k_attn<<<dim3(SS / 128, NHH, BB), 256, 0, stream>>>(qkvb, amask, ctxb);

    // out projection -> fp32 [M,H]
    k_gemm<1><<<dim3(HH / 128, M / 128), 256, 0, stream>>>(
        ctxb, woutb, bout, attnf, nullptr, HH, HH, nullptr, nullptr);

    // x1 = LN(x + attn)
    k_add_ln<<<(int)M, 256, 0, stream>>>(x, attnf, ln1g, ln1b, x1f, x1b);

    // expert routing
    k_zero3<<<1, 32, 0, stream>>>(cnt);
    k_build_idx<<<(int)((M + 255) / 256), 256, 0, stream>>>(mmod, cnt, idx, (int)M);

    // experts: gathered GEMM1 (+GELU) then GEMM2 with scatter
    for (int e = 0; e < 3; e++) {
        k_gemm<2><<<dim3(FFF / 128, M / 128), 256, 0, stream>>>(
            x1b, w1b[e], b1s[e], nullptr, hb, FFF, HH, idx + (long)e * M, cnt + e);
        k_gemm<3><<<dim3(HH / 128, M / 128), 256, 0, stream>>>(
            hb, w2b[e], b2s[e], eoutf, nullptr, HH, FFF, idx + (long)e * M, cnt + e);
    }

    // final LN -> d_out fp32
    k_add_ln<<<(int)M, 256, 0, stream>>>(x1f, eoutf, ln2g, ln2b, (float*)d_out, nullptr);
}
